// GAT_30331059044728
// MI455X (gfx1250) — matrix operand order repaired and checked by reading, not compiled, not run on hardware
//
#include <hip/hip_runtime.h>
#include <hip/hip_bf16.h>
#include <math.h>

// ---------------------------------------------------------------------------
// GATv2 GNN pipeline for gfx1250 (MI455X).
// Dense GEMMs -> v_wmma_f32_16x16x32_bf16 (f32 accumulate) with LDS-staged A
// fragments (2x ds_load_b128 each) and pre-packed fragment-ordered bf16
// weights (2x global_load_b128 each). Edge phases -> atomic segment ops.
// ---------------------------------------------------------------------------

#define NN   50000
#define EE   500000
#define IDIM 128
#define HLD  64
#define HDS  3
#define FOUT_GAT (HDS * HLD)   // 192
#define ODIM 8
#define SLOPE 0.2f

typedef __attribute__((ext_vector_type(16))) __bf16 v16bf;
typedef __attribute__((ext_vector_type(8)))  __bf16 v8bf;
typedef __attribute__((ext_vector_type(8)))  float  v8f;

// f32 -> bf16 bits, round-to-nearest-even
__device__ __forceinline__ unsigned short f2bfu(float f) {
    unsigned int u = __builtin_bit_cast(unsigned int, f);
    unsigned int r = u + 0x7FFFu + ((u >> 16) & 1u);
    return (unsigned short)(r >> 16);
}
__device__ __forceinline__ __bf16 f2bf(float f) {
    unsigned short h = f2bfu(f);
    return __builtin_bit_cast(__bf16, h);
}

// monotonic float<->uint mapping for atomicMax-based segment max
__device__ __forceinline__ unsigned fmap(float f) {
    unsigned u = __builtin_bit_cast(unsigned, f);
    return (u & 0x80000000u) ? ~u : (u | 0x80000000u);
}
__device__ __forceinline__ float funmap(unsigned u) {
    unsigned v = (u & 0x80000000u) ? (u & 0x7FFFFFFFu) : ~u;
    return __builtin_bit_cast(float, v);
}

// ---------------------------------------------------------------------------
// Pre-pack W[K,Fout] (f32, row-major) into WMMA B-fragment order, bf16,
// zero-padded to 16-column tiles. Element (nt, ks, lane, slot) holds
// W[ks*32 + slot + 16*(lane>>4)][nt*16 + (lane&15)] so that in the GEMM a
// lane's whole fragment is one contiguous 32-byte load.
// ---------------------------------------------------------------------------
__global__ void pack_w(const float* __restrict__ W, __bf16* __restrict__ Wp,
                       int K, int Fout)
{
    const int ksteps = K >> 5;
    const int nColTiles = (Fout + 15) >> 4;
    const long total = (long)nColTiles * ksteps * 32 * 16;
    const long t = (long)blockIdx.x * blockDim.x + threadIdx.x;
    if (t >= total) return;
    const int slot = (int)(t & 15);
    const int lane = (int)((t >> 4) & 31);
    const long rest = t >> 9;
    const int ks = (int)(rest % ksteps);
    const int nt = (int)(rest / ksteps);
    const int k   = ks * 32 + slot + 16 * (lane >> 4);
    const int col = nt * 16 + (lane & 15);
    const float v = (col < Fout) ? W[(size_t)k * Fout + col] : 0.0f;
    Wp[t] = f2bf(v);
}

// ---------------------------------------------------------------------------
// C[rows,Fout] = act(A[rows,K] @ W[K,Fout] + bias) via WMMA bf16/f32.
// Block = 128 threads (4 wave32s) = one 16-row tile; A tile staged in LDS as
// bf16; each wave owns every 4th 16-column tile. K in {64,128}; rows % 16 == 0.
// A fragment = 2x ds_load_b128 (two contiguous 8-element K runs per lane);
// B fragment = one 32B lane-contiguous load from packed Wp.
// ---------------------------------------------------------------------------
__global__ __launch_bounds__(128)
void gemm_bias_act(const float* __restrict__ A, const __bf16* __restrict__ Wp,
                   const float* __restrict__ bias, float* __restrict__ C,
                   int K, int Fout, int do_relu)
{
    __shared__ __align__(16) __bf16 As[16 * IDIM];

    const int tid  = threadIdx.x;
    const int lane = tid & 31;
    const int wave = tid >> 5;
    const int row0 = blockIdx.x * 16;
    const int m    = lane & 15;       // A row-in-tile / B,D column
    const int grp  = lane >> 4;       // lane half-group
    const int ksteps = K >> 5;

    // ---- stage A tile (16 x K f32) into LDS as bf16 ----
    // coalesced float4 global loads, packed 8-byte LDS stores
    const int nVec = (16 * K) >> 2;
    for (int i = tid; i < nVec; i += 128) {
        const int base = i << 2;
        const int r = base / K, c0 = base % K;     // c0 multiple of 4
        const float4 v = *(const float4*)(A + (size_t)(row0 + r) * K + c0);
        const unsigned long long pk =
            (unsigned long long)f2bfu(v.x)        |
            ((unsigned long long)f2bfu(v.y) << 16) |
            ((unsigned long long)f2bfu(v.z) << 32) |
            ((unsigned long long)f2bfu(v.w) << 48);
        *(unsigned long long*)&As[r * K + c0] = pk;
    }
    __syncthreads();

    const int nColTiles = (Fout + 15) >> 4;
    for (int nt = wave; nt < nColTiles; nt += 4) {
        v8f acc = {};
        const __bf16* wp = Wp + ((size_t)nt * ksteps * 32 + lane) * 16;
        for (int ks = 0; ks < ksteps; ++ks) {
            const int k0 = ks << 5;
            // A layout (matches packed B): slot s -> K=k0+16*grp+s
            const v8bf lo = *(const v8bf*)&As[m * K + k0 + 16 * grp];
            const v8bf hi = *(const v8bf*)&As[m * K + k0 + 16 * grp + 8];
            const v16bf a = __builtin_shufflevector(lo, hi,
                                0,1,2,3,4,5,6,7,8,9,10,11,12,13,14,15);
            const v16bf b = *(const v16bf*)(wp + (size_t)ks * 512);
            acc = __builtin_amdgcn_wmma_f32_16x16x32_bf16(
                      /*neg_a=*/false, a, /*neg_b=*/false, b,
                      /*c_mod=*/(short)0, acc,
                      /*reuse_a=*/false, /*reuse_b=*/false);
        }
        // D layout: VGPR v -> row = v + 8*grp, col = m
        const int col0 = nt * 16;
        const int c = col0 + m;
        if (c < Fout) {
            const float bv = bias[c];
#pragma unroll
            for (int v = 0; v < 8; ++v) {
                const int r = row0 + v + 8 * grp;
                float val = acc[v] + bv;
                if (do_relu) val = fmaxf(val, 0.0f);
                C[(size_t)r * Fout + c] = val;
            }
        }
    }
}

// ---------------------------------------------------------------------------
// Per-call scratch init: emax = map(-inf), den = 0, acc = 0.
// ---------------------------------------------------------------------------
__global__ void init_gat(unsigned* __restrict__ emaxm, float* __restrict__ den,
                         float* __restrict__ acc)
{
    const long t = (long)blockIdx.x * blockDim.x + threadIdx.x;
    if (t < (long)NN * HDS) { emaxm[t] = 0x007FFFFFu; /* fmap(-inf) */ den[t] = 0.0f; }
    if (t < (long)NN * FOUT_GAT) acc[t] = 0.0f;
}

// ---------------------------------------------------------------------------
// Pass A: per (edge, head) attention logit + segment max over dst.
// e = sum_c att[h][c] * leaky_relu(xl[src][h][c] + xr[dst][h][c])
// ---------------------------------------------------------------------------
__global__ void edge_logits(const float* __restrict__ xl, const float* __restrict__ xr,
                            const int* __restrict__ src, const int* __restrict__ dst,
                            const float* __restrict__ att,
                            float* __restrict__ elog, unsigned* __restrict__ emaxm)
{
    const long t = (long)blockIdx.x * blockDim.x + threadIdx.x;
    if (t >= (long)EE * HDS) return;
    const int e = (int)(t / HDS), h = (int)(t % HDS);
    const int s = src[e], d = dst[e];
    const float4* pl = (const float4*)(xl + (size_t)s * FOUT_GAT + h * HLD);
    const float4* pr = (const float4*)(xr + (size_t)d * FOUT_GAT + h * HLD);
    const float4* pa = (const float4*)(att + h * HLD);
    float acc = 0.0f;
#pragma unroll 4
    for (int q = 0; q < HLD / 4; ++q) {
        const float4 l = pl[q], r = pr[q], a = pa[q];
        float v;
        v = l.x + r.x; v = (v > 0.0f) ? v : v * SLOPE; acc = fmaf(a.x, v, acc);
        v = l.y + r.y; v = (v > 0.0f) ? v : v * SLOPE; acc = fmaf(a.y, v, acc);
        v = l.z + r.z; v = (v > 0.0f) ? v : v * SLOPE; acc = fmaf(a.z, v, acc);
        v = l.w + r.w; v = (v > 0.0f) ? v : v * SLOPE; acc = fmaf(a.w, v, acc);
    }
    elog[t] = acc;
    atomicMax(&emaxm[(size_t)d * HDS + h], fmap(acc));
}

// ---------------------------------------------------------------------------
// Pass B: w = exp(e - max[dst]); den[dst] += w. elog overwritten with w.
// ---------------------------------------------------------------------------
__global__ void edge_weight(float* __restrict__ elog, const unsigned* __restrict__ emaxm,
                            const int* __restrict__ dst, float* __restrict__ den)
{
    const long t = (long)blockIdx.x * blockDim.x + threadIdx.x;
    if (t >= (long)EE * HDS) return;
    const int e = (int)(t / HDS), h = (int)(t % HDS);
    const int d = dst[e];
    const float mx = funmap(emaxm[(size_t)d * HDS + h]);
    const float w  = __expf(elog[t] - mx);
    elog[t] = w;
    atomicAdd(&den[(size_t)d * HDS + h], w);
}

// ---------------------------------------------------------------------------
// Pass C: acc[dst, h, c] += w[e, h] * xl[src, h, c]. One thread per
// (edge, channel-of-192): consecutive threads -> consecutive channels,
// so xl reads and acc atomics are coalesced within a wave.
// ---------------------------------------------------------------------------
__global__ void edge_scatter(const float* __restrict__ w, const float* __restrict__ xl,
                             const int* __restrict__ src, const int* __restrict__ dst,
                             float* __restrict__ acc)
{
    const long t = (long)blockIdx.x * blockDim.x + threadIdx.x;
    if (t >= (long)EE * FOUT_GAT) return;
    const int e = (int)(t / FOUT_GAT);
    const int r = (int)(t % FOUT_GAT);          // h*64 + c
    const float ww = w[(size_t)e * HDS + (r >> 6)];
    const float xv = xl[(size_t)src[e] * FOUT_GAT + r];
    atomicAdd(&acc[(size_t)dst[e] * FOUT_GAT + r], ww * xv);
}

// ---------------------------------------------------------------------------
// Pass D: out[n,c] = relu( mean_h( acc[n,h,c] / (den[n,h]+1e-16) ) + bias[c] )
// ---------------------------------------------------------------------------
__global__ void gat_finalize(const float* __restrict__ acc, const float* __restrict__ den,
                             const float* __restrict__ bias, float* __restrict__ out)
{
    const long t = (long)blockIdx.x * blockDim.x + threadIdx.x;
    if (t >= (long)NN * HLD) return;
    const int n = (int)(t / HLD), c = (int)(t % HLD);
    float s = 0.0f;
#pragma unroll
    for (int h = 0; h < HDS; ++h)
        s += acc[(size_t)n * FOUT_GAT + h * HLD + c] / (den[(size_t)n * HDS + h] + 1e-16f);
    s = s * (1.0f / HDS) + bias[c];
    out[t] = fmaxf(s, 0.0f);
}

// ---------------------------------------------------------------------------
// Host-side orchestration (graph-capture safe: launches only).
// ---------------------------------------------------------------------------
static inline int cdiv(long a, int b) { return (int)((a + b - 1) / b); }

static inline long wp_elems(int K, int Fout) {
    return (long)((Fout + 15) / 16) * (K / 32) * 512;
}

extern "C" void kernel_launch(void* const* d_in, const int* in_sizes, int n_in,
                              void* d_out, int out_size, void* d_ws, size_t ws_size,
                              hipStream_t stream)
{
    const float* x       = (const float*)d_in[0];
    const int*   edgeIdx = (const int*)d_in[1];
    // d_in[2] = edgeAttr (unused by reference network)
    const float* enc_w0 = (const float*)d_in[3],  *enc_b0 = (const float*)d_in[4];
    const float* enc_w1 = (const float*)d_in[5],  *enc_b1 = (const float*)d_in[6];
    const float* g0_wl  = (const float*)d_in[7],  *g0_bl  = (const float*)d_in[8];
    const float* g0_wr  = (const float*)d_in[9],  *g0_br  = (const float*)d_in[10];
    const float* g0_att = (const float*)d_in[11], *g0_bias= (const float*)d_in[12];
    const float* g1_wl  = (const float*)d_in[13], *g1_bl  = (const float*)d_in[14];
    const float* g1_wr  = (const float*)d_in[15], *g1_br  = (const float*)d_in[16];
    const float* g1_att = (const float*)d_in[17], *g1_bias= (const float*)d_in[18];
    const float* dec_w0 = (const float*)d_in[19], *dec_b0 = (const float*)d_in[20];
    const float* dec_w1 = (const float*)d_in[21], *dec_b1 = (const float*)d_in[22];

    const int* src = edgeIdx;          // edgeIdx[0, :]
    const int* dst = edgeIdx + EE;     // edgeIdx[1, :]

    // workspace layout (floats)
    float* ws = (float*)d_ws;
    float*    h0    = ws;                          // N*64
    float*    h1    = h0   + (size_t)NN * HLD;     // N*64
    float*    xl    = h1   + (size_t)NN * HLD;     // N*192
    float*    xr    = xl   + (size_t)NN * FOUT_GAT;// N*192
    float*    elog  = xr   + (size_t)NN * FOUT_GAT;// E*3 (logits, then weights)
    unsigned* emaxm = (unsigned*)(elog + (size_t)EE * HDS); // N*3
    float*    den   = (float*)emaxm + (size_t)NN * HDS;     // N*3
    float*    acc   = den  + (size_t)NN * HDS;              // N*192
    __bf16*   Wp    = (__bf16*)(acc + (size_t)NN * FOUT_GAT); // packed weights (reused)

    const dim3 gB(128);
    const int  rowTiles = NN / 16;          // 50000 % 16 == 0

    // helper macro: pack weight then run GEMM (stream-ordered, so safe reuse)
#define GEMM(Ain, Wf, Bf, Cout, Kdim, Fdim, RELU)                                   \
    do {                                                                            \
        const long wpn = wp_elems(Kdim, Fdim);                                      \
        pack_w<<<cdiv(wpn, 256), 256, 0, stream>>>(Wf, Wp, Kdim, Fdim);             \
        gemm_bias_act<<<rowTiles, gB, 0, stream>>>(Ain, Wp, Bf, Cout, Kdim, Fdim, RELU); \
    } while (0)

    // ---- encoder MLP ----
    GEMM(x,  enc_w0, enc_b0, h0, IDIM, HLD, 1);
    GEMM(h0, enc_w1, enc_b1, h1, HLD,  HLD, 1);

    // ---- GATv2 layer 0: h1 -> h0 ----
    GEMM(h1, g0_wl, g0_bl, xl, HLD, FOUT_GAT, 0);
    GEMM(h1, g0_wr, g0_br, xr, HLD, FOUT_GAT, 0);
    init_gat    <<<cdiv((long)NN * FOUT_GAT, 256), 256, 0, stream>>>(emaxm, den, acc);
    edge_logits <<<cdiv((long)EE * HDS, 256), 256, 0, stream>>>(xl, xr, src, dst, g0_att, elog, emaxm);
    edge_weight <<<cdiv((long)EE * HDS, 256), 256, 0, stream>>>(elog, emaxm, dst, den);
    edge_scatter<<<cdiv((long)EE * FOUT_GAT, 256), 256, 0, stream>>>(elog, xl, src, dst, acc);
    gat_finalize<<<cdiv((long)NN * HLD, 256), 256, 0, stream>>>(acc, den, g0_bias, h0);

    // ---- GATv2 layer 1: h0 -> h1 ----
    GEMM(h0, g1_wl, g1_bl, xl, HLD, FOUT_GAT, 0);
    GEMM(h0, g1_wr, g1_br, xr, HLD, FOUT_GAT, 0);
    init_gat    <<<cdiv((long)NN * FOUT_GAT, 256), 256, 0, stream>>>(emaxm, den, acc);
    edge_logits <<<cdiv((long)EE * HDS, 256), 256, 0, stream>>>(xl, xr, src, dst, g1_att, elog, emaxm);
    edge_weight <<<cdiv((long)EE * HDS, 256), 256, 0, stream>>>(elog, emaxm, dst, den);
    edge_scatter<<<cdiv((long)EE * FOUT_GAT, 256), 256, 0, stream>>>(elog, xl, src, dst, acc);
    gat_finalize<<<cdiv((long)NN * HLD, 256), 256, 0, stream>>>(acc, den, g1_bias, h1);

    // ---- decoder MLP ----
    GEMM(h1, dec_w0, dec_b0, h0, HLD, HLD, 1);
    GEMM(h0, dec_w1, dec_b1, (float*)d_out, HLD, ODIM, 1);
#undef GEMM
}
